// BPN_89833535964043
// MI455X (gfx1250) — matrix operand order repaired and checked by reading, not compiled, not run on hardware
//
#include <hip/hip_runtime.h>

#define NNODES 100000
#define NEDGES 1600000
#define FDIM 16
#define HDIM 128

typedef __attribute__((ext_vector_type(16))) _Float16 v16h;
typedef __attribute__((ext_vector_type(8)))  float    v8f;

__device__ __forceinline__ float lrelu(float x, float s){ return x >= 0.f ? x : s*x; }

// A fragment (16xK f16, row-major LDS tile, stride ldk halves), per CDNA5 ISA:
// lanes 0-15: row=lane, elems 0..7 -> k0..k0+7, elems 8..15 -> k0+16..k0+23
// lanes 16-31: row=lane-16, k offset +8 within each half.
__device__ __forceinline__ v16h load_a_frag(const _Float16* tile, int ldk, int k0, int lane){
  int row = lane & 15;
  int khalf = (lane >> 4) << 3;                 // 0 or 8
  const _Float16* p0 = tile + row*ldk + k0 + khalf;
  v16h a;
  #pragma unroll
  for (int i=0;i<8;i++){ a[i]=p0[i]; a[8+i]=p0[16+i]; }
  return a;
}

// B fragment (Kx16), LDS tile stored transposed: bt[col][k], stride ldk halves.
// lanes 0-15: col=lane, elems 0..7 -> k0..k0+7, elems 8..15 -> k0+16..k0+23 ; lanes 16-31: k offset +8.
__device__ __forceinline__ v16h load_b_frag(const _Float16* bt, int ldk, int k0, int lane){
  int col = lane & 15;
  int koff = k0 + ((lane >> 4) << 3);           // 0 or 8
  const _Float16* p = bt + col*ldk + koff;
  v16h b;
  #pragma unroll
  for (int i=0;i<8;i++){ b[i]=p[i]; b[8+i]=p[16+i]; }
  return b;
}

// monotonic float atomic max via signed/unsigned int atomics (init memory to -inf-ish)
__device__ __forceinline__ void atomic_max_f32(float* addr, float val){
  if (val >= 0.f) atomicMax((int*)addr, __float_as_int(val));
  else            atomicMin((unsigned int*)addr, (unsigned int)__float_as_int(val));
}

// ---------------- tiny: c_bit = dot(W_nm[128,:], attn) ----------------
__global__ void k_cbit(const float* __restrict__ Wnm, const float* __restrict__ attn,
                       float* __restrict__ cbit){
  __shared__ float red[128];
  int t = threadIdx.x;
  red[t] = Wnm[128*128 + t] * attn[t];
  __syncthreads();
  for (int s = 64; s > 0; s >>= 1){ if (t < s) red[t] += red[t+s]; __syncthreads(); }
  if (t == 0) cbit[0] = red[0];
}

// ---------------- init scratch ----------------
__global__ void k_init(float* __restrict__ neigh, float* __restrict__ emax,
                       float* __restrict__ esum, float* __restrict__ sbit){
  int i = blockIdx.x*blockDim.x + threadIdx.x;
  if (i < NNODES*HDIM) neigh[i] = 0.f;
  if (i < NNODES){ emax[i] = -1e30f; esum[i] = 0.f; sbit[i] = 0.f; }
}

// ---------------- per 16-node tile: h = mlp_self(feat); P = h@Wnm_h; Q' = feat@Wnm_f + bnm
__global__ void __launch_bounds__(256) k_node_embed(
    const float* __restrict__ feat, const float* __restrict__ W1, const float* __restrict__ b1,
    const float* __restrict__ W2,   const float* __restrict__ b2,
    const float* __restrict__ Wnm,  const float* __restrict__ bnm,
    float* __restrict__ P, float* __restrict__ Q)
{
  extern __shared__ _Float16 sm[];
  _Float16* featA = sm;                  // 16x32 (K padded 16->32)
  _Float16* W1T   = featA + 16*32;       // 64 cols x 32 k
  _Float16* midA  = W1T   + 64*32;       // 16x64
  _Float16* W2T   = midA  + 16*64;       // 128 cols x 64 k
  _Float16* hA    = W2T   + 128*64;      // 16x128
  _Float16* WnmT  = hA    + 16*128;      // 128 cols x 128 k
  _Float16* WqT   = WnmT  + 128*128;     // 128 cols x 32 k (pad 16->32)

  int tid = threadIdx.x;
  int node0 = blockIdx.x * 16;

  for (int i = tid; i < 16*32;   i += 256){ int r=i>>5,k=i&31;  featA[i]=(k<FDIM)?(_Float16)feat[(node0+r)*FDIM+k]:(_Float16)0.f; }
  for (int i = tid; i < 64*32;   i += 256){ int c=i>>5,k=i&31;  W1T[i]  =(k<FDIM)?(_Float16)W1[k*64+c]:(_Float16)0.f; }
  for (int i = tid; i < 128*64;  i += 256){ int c=i>>6,k=i&63;  W2T[i]  =(_Float16)W2[k*128+c]; }
  for (int i = tid; i < 128*128; i += 256){ int c=i>>7,k=i&127; WnmT[i] =(_Float16)Wnm[k*128+c]; }
  for (int i = tid; i < 128*32;  i += 256){ int c=i>>5,k=i&31;  WqT[i]  =(k<FDIM)?(_Float16)Wnm[(129+k)*128+c]:(_Float16)0.f; }
  __syncthreads();

  int w = tid >> 5, lane = tid & 31;
  int n = lane & 15, mb = (lane >> 4) << 3;

  if (w < 4){                                    // layer1: [16x16]@[16x64], K zero-padded
    v8f c = {};
    v16h a = load_a_frag(featA, 32, 0, lane);
    v16h b = load_b_frag(W1T + (w*16)*32, 32, 0, lane);
    c = __builtin_amdgcn_wmma_f32_16x16x32_f16(false, a, false, b, (short)0, c, false, false);
    float bias = b1[w*16+n];
    #pragma unroll
    for (int v=0; v<8; v++) midA[(mb+v)*64 + w*16 + n] = (_Float16)lrelu(c[v]+bias, 0.1f);
  }
  __syncthreads();

  {                                              // layer2: [16x64]@[64x128]
    v8f c = {};
    #pragma unroll
    for (int kk=0; kk<64; kk+=32){
      v16h a = load_a_frag(midA, 64, kk, lane);
      v16h b = load_b_frag(W2T + (w*16)*64, 64, kk, lane);
      c = __builtin_amdgcn_wmma_f32_16x16x32_f16(false, a, false, b, (short)0, c, false, false);
    }
    float bias = b2[w*16+n];
    #pragma unroll
    for (int v=0; v<8; v++) hA[(mb+v)*128 + w*16 + n] = (_Float16)(c[v]+bias);
  }
  __syncthreads();

  {                                              // P = h @ Wnm[0:128,:]
    v8f c = {};
    #pragma unroll
    for (int kk=0; kk<128; kk+=32){
      v16h a = load_a_frag(hA, 128, kk, lane);
      v16h b = load_b_frag(WnmT + (w*16)*128, 128, kk, lane);
      c = __builtin_amdgcn_wmma_f32_16x16x32_f16(false, a, false, b, (short)0, c, false, false);
    }
    #pragma unroll
    for (int v=0; v<8; v++) P[(size_t)(node0+mb+v)*HDIM + w*16 + n] = c[v];
  }

  {                                              // Q' = feat @ Wnm[129:145,:] + bnm
    v8f c = {};
    v16h a = load_a_frag(featA, 32, 0, lane);
    v16h b = load_b_frag(WqT + (w*16)*32, 32, 0, lane);
    c = __builtin_amdgcn_wmma_f32_16x16x32_f16(false, a, false, b, (short)0, c, false, false);
    float bias = bnm[w*16+n];
    #pragma unroll
    for (int v=0; v<8; v++) Q[(size_t)(node0+mb+v)*HDIM + w*16 + n] = c[v] + bias;
  }
}

// ---------------- pa = P.attn, qa = Q'.attn (wave per node) ----------------
__global__ void __launch_bounds__(256) k_attn_node(const float* __restrict__ P, const float* __restrict__ Q,
    const float* __restrict__ attn, float* __restrict__ pa, float* __restrict__ qa){
  int gid = blockIdx.x*256 + threadIdx.x;
  int node = gid >> 5, lane = gid & 31;
  if (node >= NNODES) return;
  float4 av = ((const float4*)attn)[lane];
  float4 pv = ((const float4*)P)[(size_t)node*32 + lane];
  float4 qv = ((const float4*)Q)[(size_t)node*32 + lane];
  float sp = pv.x*av.x + pv.y*av.y + pv.z*av.z + pv.w*av.w;
  float sq = qv.x*av.x + qv.y*av.y + qv.z*av.z + qv.w*av.w;
  #pragma unroll
  for (int m=16; m>0; m>>=1){ sp += __shfl_xor(sp, m); sq += __shfl_xor(sq, m); }
  if (lane == 0){ pa[node] = sp; qa[node] = sq; }
}

// ---------------- edge pass 1: logits + segment max ----------------
__global__ void k_edge_logit(const int* __restrict__ src, const int* __restrict__ dst,
    const float* __restrict__ bit, const float* __restrict__ pa, const float* __restrict__ qa,
    const float* __restrict__ cbit, float* __restrict__ earr, float* __restrict__ emax){
  int e = blockIdx.x*blockDim.x + threadIdx.x;
  if (e >= NEDGES) return;
  int s = src[e], d = dst[e];
  float ev = lrelu(pa[s] + qa[d] + bit[e]*cbit[0], 0.2f);
  earr[e] = ev;
  atomic_max_f32(&emax[d], ev);
}

// ---------------- edge pass 2: exp + segment sum ----------------
__global__ void k_edge_exp(const int* __restrict__ dst, float* __restrict__ earr,
    const float* __restrict__ emax, float* __restrict__ esum){
  int e = blockIdx.x*blockDim.x + threadIdx.x;
  if (e >= NEDGES) return;
  int d = dst[e];
  float ee = __expf(earr[e] - emax[d]);
  earr[e] = ee;
  atomicAdd(&esum[d], ee);
}

// ---------------- edge pass 3: neigh += alpha*P[src]; sbit += alpha*bit (wave per edge)
__global__ void __launch_bounds__(256) k_edge_scatter(const int* __restrict__ src, const int* __restrict__ dst,
    const float* __restrict__ bit, const float* __restrict__ earr, const float* __restrict__ esum,
    const float* __restrict__ P, float* __restrict__ neigh, float* __restrict__ sbit){
  int gid = blockIdx.x*256 + threadIdx.x;
  int e = gid >> 5, lane = gid & 31;
  if (e >= NEDGES) return;
  int s = src[e], d = dst[e];
  float alpha = earr[e] / esum[d];
  const float4* prow = (const float4*)(P + (size_t)s*HDIM);
  __builtin_prefetch(prow + lane, 0, 0);
  float4 pv = prow[lane];
  float* nrow = neigh + (size_t)d*HDIM + lane*4;
  atomicAdd(nrow+0, alpha*pv.x);
  atomicAdd(nrow+1, alpha*pv.y);
  atomicAdd(nrow+2, alpha*pv.z);
  atomicAdd(nrow+3, alpha*pv.w);
  if (lane == 0) atomicAdd(&sbit[d], alpha*bit[e]);
}

// ---------------- per 16-node tile: out = lrelu(relu(neigh_full)@Wo1+bo1,0.1)@Wo2+bo2
__global__ void __launch_bounds__(256) k_node_out(const float* __restrict__ neigh,
    const float* __restrict__ Q, const float* __restrict__ sbit, const float* __restrict__ esum,
    const float* __restrict__ Wnm, const float* __restrict__ Wo1, const float* __restrict__ bo1,
    const float* __restrict__ Wo2, const float* __restrict__ bo2, float* __restrict__ out){
  extern __shared__ _Float16 sm[];
  _Float16* tA  = sm;                       // 16x128
  _Float16* WoT = tA + 16*128;              // 128 cols x 128 k
  float* outacc = (float*)(WoT + 128*128);  // 16

  int tid = threadIdx.x;
  int node0 = blockIdx.x * 16;

  for (int i = tid; i < 16*128; i += 256){
    int r = i >> 7, j = i & 127;
    int node = node0 + r;
    float has = (esum[node] > 0.f) ? 1.f : 0.f;   // nodes with no in-edges: neigh==0 exactly
    float val = neigh[(size_t)node*HDIM + j]
              + has*(Q[(size_t)node*HDIM + j] + Wnm[128*128 + j]*sbit[node]);
    tA[i] = (_Float16)fmaxf(val, 0.f);
  }
  for (int i = tid; i < 128*128; i += 256){ int c=i>>7,k=i&127; WoT[i]=(_Float16)Wo1[k*128+c]; }
  if (tid < 16) outacc[tid] = 0.f;
  __syncthreads();

  int w = tid >> 5, lane = tid & 31;
  int n = lane & 15, mb = (lane >> 4) << 3;
  v8f c = {};
  #pragma unroll
  for (int kk=0; kk<128; kk+=32){
    v16h a = load_a_frag(tA, 128, kk, lane);
    v16h b = load_b_frag(WoT + (w*16)*128, 128, kk, lane);
    c = __builtin_amdgcn_wmma_f32_16x16x32_f16(false, a, false, b, (short)0, c, false, false);
  }
  int col = w*16 + n;
  float bias = bo1[col];
  float wv = Wo2[col];
  float ps[8];
  #pragma unroll
  for (int v=0; v<8; v++) ps[v] = lrelu(c[v]+bias, 0.1f) * wv;
  #pragma unroll
  for (int m=8; m>0; m>>=1){
    #pragma unroll
    for (int v=0; v<8; v++) ps[v] += __shfl_xor(ps[v], m);
  }
  if ((lane & 15) == 0){
    #pragma unroll
    for (int v=0; v<8; v++) atomicAdd(&outacc[mb+v], ps[v]);
  }
  __syncthreads();
  if (tid < 16) out[node0 + tid] = outacc[tid] + bo2[0];
}

extern "C" void kernel_launch(void* const* d_in, const int* in_sizes, int n_in,
                              void* d_out, int out_size, void* d_ws, size_t ws_size,
                              hipStream_t stream){
  const float* feat = (const float*)d_in[0];
  const float* bit  = (const float*)d_in[1];
  const int*   src  = (const int*)d_in[2];
  const int*   dst  = (const int*)d_in[3];
  const float* W1   = (const float*)d_in[4];
  const float* b1   = (const float*)d_in[5];
  const float* W2   = (const float*)d_in[6];
  const float* b2   = (const float*)d_in[7];
  const float* Wnm  = (const float*)d_in[8];
  const float* bnm  = (const float*)d_in[9];
  const float* attn = (const float*)d_in[10];
  const float* Wo1  = (const float*)d_in[11];
  const float* bo1  = (const float*)d_in[12];
  const float* Wo2  = (const float*)d_in[13];
  const float* bo2  = (const float*)d_in[14];
  float* out = (float*)d_out;

  float* W = (float*)d_ws;
  size_t o = 0;
  float* P     = W + o; o += (size_t)NNODES*HDIM;
  float* Q     = W + o; o += (size_t)NNODES*HDIM;
  float* neigh = W + o; o += (size_t)NNODES*HDIM;
  float* pa    = W + o; o += NNODES;
  float* qa    = W + o; o += NNODES;
  float* emax  = W + o; o += NNODES;
  float* esum  = W + o; o += NNODES;
  float* sbit  = W + o; o += NNODES;
  float* earr  = W + o; o += NEDGES;
  float* cbit  = W + o; o += 1;

  size_t lds1 = (size_t)(16*32 + 64*32 + 16*64 + 128*64 + 16*128 + 128*128 + 128*32)*sizeof(_Float16);
  size_t lds6 = (size_t)(16*128 + 128*128)*sizeof(_Float16) + 16*sizeof(float);

  k_cbit<<<1, 128, 0, stream>>>(Wnm, attn, cbit);
  k_init<<<(NNODES*HDIM + 255)/256, 256, 0, stream>>>(neigh, emax, esum, sbit);
  k_node_embed<<<NNODES/16, 256, lds1, stream>>>(feat, W1, b1, W2, b2, Wnm, bnm, P, Q);
  k_attn_node<<<(NNODES*32 + 255)/256, 256, 0, stream>>>(P, Q, attn, pa, qa);
  k_edge_logit<<<(NEDGES + 255)/256, 256, 0, stream>>>(src, dst, bit, pa, qa, cbit, earr, emax);
  k_edge_exp<<<(NEDGES + 255)/256, 256, 0, stream>>>(dst, earr, emax, esum);
  k_edge_scatter<<<(NEDGES*32 + 255)/256, 256, 0, stream>>>(src, dst, bit, earr, esum, P, neigh, sbit);
  k_node_out<<<NNODES/16, 256, lds6, stream>>>(neigh, Q, sbit, esum, Wnm, Wo1, bo1, Wo2, bo2, out);
}
